// Model_19456201851461
// MI455X (gfx1250) — compile-verified, matrix operand order checked by reading, not run on hardware
//
#include <hip/hip_runtime.h>
#include <hip/hip_bf16.h>

typedef __attribute__((ext_vector_type(16))) _Float16 v16h;
typedef __attribute__((ext_vector_type(8)))  _Float16 v8h;
typedef __attribute__((ext_vector_type(8)))  float    v8f;

// Problem constants
#define NB_   4
#define B_    256
#define L_    512
#define C_    32
#define SEG_  64
#define FREQ_ 8
#define STEP_ 4
#define D_    512
#define H_    1024
#define FIN_  2048
#define F_    257
#define BC_   (B_ * C_)         // 8192
#define EPS_  1e-5f

// s_wait_asynccnt: builtin when available, else inline asm
#if __has_builtin(__builtin_amdgcn_s_wait_asynccnt)
#define WAIT_ASYNC(n) __builtin_amdgcn_s_wait_asynccnt(n)
#else
#define WAIT_ASYNC(n) asm volatile("s_wait_asynccnt %0" :: "i"(n) : "memory")
#endif

// ---------------------------------------------------------------------------
// Instance norm over time: mean/std per (b,c), write normalized f16 [BC, L]
// ---------------------------------------------------------------------------
__global__ __launch_bounds__(256)
void k_instnorm(const float* __restrict__ x, float* __restrict__ meanv,
                float* __restrict__ stdv, _Float16* __restrict__ xn2) {
  int w    = blockIdx.x * 8 + (threadIdx.x >> 5);   // 0..BC-1 == b*32+c
  int lane = threadIdx.x & 31;
  int b = w >> 5, c = w & 31;
  const float* xp = x + (long)b * L_ * C_ + c;
  float vals[16];
  float s = 0.f, s2 = 0.f;
  #pragma unroll
  for (int i = 0; i < 16; ++i) {
    float v = xp[(long)(lane + i * 32) * C_];
    vals[i] = v; s += v; s2 += v * v;
  }
  #pragma unroll
  for (int off = 16; off > 0; off >>= 1) {
    s  += __shfl_xor(s,  off, 32);
    s2 += __shfl_xor(s2, off, 32);
  }
  float mu  = s * (1.f / L_);
  float var = s2 * (1.f / L_) - mu * mu;
  float sd  = sqrtf(var + EPS_);
  if (lane == 0) { meanv[w] = mu; stdv[w] = sd; }
  float inv = 1.f / sd;
  _Float16* op = xn2 + (long)w * L_;
  #pragma unroll
  for (int i = 0; i < 16; ++i)
    op[lane + i * 32] = (_Float16)((vals[i] - mu) * inv);
}

// ---------------------------------------------------------------------------
// Filter taps: Kc[nb][t] = irfft(sigmoid(mask_weights[nb]))[t]
// ---------------------------------------------------------------------------
__global__ void k_build_filter(const float* __restrict__ mw, float* __restrict__ Kc) {
  int idx = blockIdx.x * blockDim.x + threadIdx.x;
  if (idx >= NB_ * L_) return;
  int nb = idx >> 9, t = idx & (L_ - 1);
  const float* m = mw + nb * F_;
  const float twopi_over_L = 6.283185307179586f / (float)L_;
  float acc = 1.f / (1.f + expf(-m[0]));
  for (int f = 1; f < F_ - 1; ++f) {
    float sg  = 1.f / (1.f + expf(-m[f]));
    int   ang = (t * f) & (L_ - 1);               // (t*f) mod L
    acc += 2.f * sg * cosf(twopi_over_L * (float)ang);
  }
  float sN = 1.f / (1.f + expf(-m[F_ - 1]));
  acc += sN * ((t & 1) ? -1.f : 1.f);             // cos(pi*t)
  Kc[idx] = acc * (1.f / (float)L_);
}

// Circulant matrices Mc[nb][t][t'] = Kc[nb][(t-t') mod L], f16
__global__ void k_build_circ(const float* __restrict__ Kc, _Float16* __restrict__ Mc) {
  long i = (long)blockIdx.x * 256 + threadIdx.x;
  if (i >= (long)NB_ * L_ * L_) return;
  int nb  = (int)(i >> 18);
  int rem = (int)(i & (L_ * L_ - 1));
  int t = rem >> 9, tp = rem & (L_ - 1);
  Mc[i] = (_Float16)Kc[nb * L_ + ((t - tp) & (L_ - 1))];
}

// ---------------------------------------------------------------------------
// Generic f32 -> f16 conversion
// ---------------------------------------------------------------------------
__global__ void k_cvt16(const float* __restrict__ src, _Float16* __restrict__ dst, long n) {
  long i = (long)blockIdx.x * 256 + threadIdx.x;
  if (i < n) dst[i] = (_Float16)src[i];
}

// Strided slice convert: dst[r*512+c] = src[r*src_ld + c]   (cols fixed = 512)
__global__ void k_cvt16_rows512(const float* __restrict__ src, _Float16* __restrict__ dst,
                                long n, long src_ld) {
  long i = (long)blockIdx.x * 256 + threadIdx.x;
  if (i >= n) return;
  long r = i >> 9, c = i & 511;
  dst[i] = (_Float16)src[r * src_ld + c];
}

// Repack filtered signal Xi f32 [L, BC] (rows t=f*64+p, cols b*32+c)
// into res16 f16 [B*FREQ, SEG*C] (rows b*8+f, cols p*32+c)
__global__ void k_pack_res(const float* __restrict__ Xi, _Float16* __restrict__ res) {
  long i = (long)blockIdx.x * 256 + threadIdx.x;   // over 2048*2048
  if (i >= (long)(B_ * FREQ_) * FIN_) return;
  int col = (int)(i & (FIN_ - 1)), row = (int)(i >> 11);
  int b = row >> 3, f = row & 7;
  int p = col >> 5, c = col & 31;
  res[i] = (_Float16)Xi[(long)(f * SEG_ + p) * BC_ + b * C_ + c];
}

// ---------------------------------------------------------------------------
// WMMA GEMM: C[M,N](f32) = A[M,K](f16,row-major) x Bw[N,K](f16,row-major)^T
// + optional bias[N], ReLU, aux[row*auxld+col] add; dual f32/f16 stores.
//
// Block = 128(M) x 256(N) tile, 256 threads = 8 wave32 arranged 2(M) x 4(N);
// wave tile = 64x64 = 4x4 WMMA accumulators (16 ds_load_b128 per 16 WMMA).
// K staged in 32-wide double-buffered LDS tiles filled with
// GLOBAL_LOAD_ASYNC_TO_LDS_B128 (ASYNCcnt), issued one stage ahead.
// Grid: (N/256, M/128). K multiple of 32. EXEC uniform (WMMA requirement).
// ---------------------------------------------------------------------------
#define AP_     40                   // padded LDS pitch in halfs (80B, 16B-aligned)
#define ATILE_  (128 * AP_)          // halfs per A tile (5120)
#define BTILE_  (256 * AP_)          // halfs per B tile (10240)
#define STAGE_  (ATILE_ + BTILE_)    // halfs per stage (15360 -> 30720 B)

__global__ __launch_bounds__(256)
void k_gemm_wmma(const _Float16* __restrict__ A, int lda,
                 const _Float16* __restrict__ Bw, int ldb,
                 float* __restrict__ Cf, int ldc,
                 _Float16* __restrict__ C16, int ldc16,
                 const float* __restrict__ bias,
                 const float* __restrict__ aux, long auxld,
                 int K, int relu) {
  __shared__ __align__(16) _Float16 smem[2 * STAGE_];   // 60 KB, double buffered

  const int bm   = blockIdx.y * 128;
  const int bn   = blockIdx.x * 256;
  const int tid  = threadIdx.x;
  const int wave = tid >> 5;
  const int lane = tid & 31;
  const int wm   = wave >> 2;        // 0..1 -> M offset wm*64
  const int wn   = wave & 3;         // 0..3 -> N offset wn*64
  const int rsel  = lane & 15;
  const int khalf = lane >> 4;

  // LDS byte address base for async-to-LDS (low 32 bits of shared aperture ptr
  // == wave-relative LDS offset per ISA flat-address mapping)
  const unsigned int sbase = (unsigned int)(size_t)&smem[0];

  v8f acc[4][4];
  #pragma unroll
  for (int mt = 0; mt < 4; ++mt)
    #pragma unroll
    for (int nt = 0; nt < 4; ++nt)
      acc[mt][nt] = (v8f)(0.f);

  const int nk = K >> 5;

  // Staging: A = 512 b128 chunks (2/thread), B = 1024 chunks (4/thread).
#define ISSUE_STAGE(st, k0)                                                     \
  {                                                                             \
    const int base = (st) * STAGE_;                                             \
    _Pragma("unroll")                                                           \
    for (int i = 0; i < 2; ++i) {                                               \
      int idx = tid + i * 256;               /* 0..511 */                       \
      int r = idx >> 2, c8 = (idx & 3) * 8;                                     \
      const _Float16* ga = &A[(long)(bm + r) * lda + (k0) + c8];                \
      unsigned int la = sbase + (unsigned int)(base + r * AP_ + c8) * 2u;       \
      asm volatile("global_load_async_to_lds_b128 %0, %1, off"                  \
                   :: "v"(la), "v"(ga) : "memory");                             \
    }                                                                           \
    _Pragma("unroll")                                                           \
    for (int i = 0; i < 4; ++i) {                                               \
      int idx = tid + i * 256;               /* 0..1023 */                      \
      int r = idx >> 2, c8 = (idx & 3) * 8;                                     \
      const _Float16* gb = &Bw[(long)(bn + r) * ldb + (k0) + c8];               \
      unsigned int lb = sbase +                                                 \
          (unsigned int)(base + ATILE_ + r * AP_ + c8) * 2u;                    \
      asm volatile("global_load_async_to_lds_b128 %0, %1, off"                  \
                   :: "v"(lb), "v"(gb) : "memory");                             \
    }                                                                           \
  }

  ISSUE_STAGE(0, 0);

  for (int kb = 0; kb < nk; ++kb) {
    const int st = kb & 1;
    if (kb + 1 < nk) {
      ISSUE_STAGE((kb + 1) & 1, (kb + 1) * 32);
      WAIT_ASYNC(6);                 // stage kb landed (in-order), kb+1 in flight
    } else {
      WAIT_ASYNC(0);
    }
    __syncthreads();

    const int abase = st * STAGE_;
    const int bbase = st * STAGE_ + ATILE_;

    // Fragment loads per ISA 16-bit A/B layout:
    // lane(0..15)=row, khalf selects K {0..7,16..23} vs {8..15,24..31}.
    v16h bfrag[4];
    #pragma unroll
    for (int nt = 0; nt < 4; ++nt) {
      int r = wn * 64 + nt * 16 + rsel;
      v8h lo = *(const v8h*)&smem[bbase + r * AP_ + khalf * 8];
      v8h hi = *(const v8h*)&smem[bbase + r * AP_ + 16 + khalf * 8];
      bfrag[nt] = __builtin_shufflevector(lo, hi, 0,1,2,3,4,5,6,7,8,9,10,11,12,13,14,15);
    }
    #pragma unroll
    for (int mt = 0; mt < 4; ++mt) {
      int r = wm * 64 + mt * 16 + rsel;
      v8h lo = *(const v8h*)&smem[abase + r * AP_ + khalf * 8];
      v8h hi = *(const v8h*)&smem[abase + r * AP_ + 16 + khalf * 8];
      v16h afrag = __builtin_shufflevector(lo, hi, 0,1,2,3,4,5,6,7,8,9,10,11,12,13,14,15);
      #pragma unroll
      for (int nt = 0; nt < 4; ++nt)
        acc[mt][nt] = __builtin_amdgcn_wmma_f32_16x16x32_f16(
            false, afrag, false, bfrag[nt],
            (short)0, acc[mt][nt], false, false);
    }

    __syncthreads();   // drain reads before this buffer is re-filled
  }
#undef ISSUE_STAGE

  // Epilogue. C/D layout: VGPR r -> M = r + 8*(lane>=16), N = lane&15.
  const int nlo = lane & 15;
  const int mhi = (lane >> 4) * 8;
  #pragma unroll
  for (int mt = 0; mt < 4; ++mt) {
    #pragma unroll
    for (int nt = 0; nt < 4; ++nt) {
      int col  = bn + wn * 64 + nt * 16 + nlo;
      int rowb = bm + wm * 64 + mt * 16 + mhi;
      float bv = bias ? bias[col] : 0.f;
      #pragma unroll
      for (int r = 0; r < 8; ++r) {
        int row = rowb + r;
        float v = acc[mt][nt][r] + bv;
        if (relu) v = v > 0.f ? v : 0.f;
        if (aux)  v += aux[(long)row * auxld + col];
        if (Cf)   Cf[(long)row * ldc + col] = v;
        if (C16)  C16[(long)row * ldc16 + col] = (_Float16)v;
      }
    }
  }
}

// ---------------------------------------------------------------------------
// LayerNorm over D=512 per row; rows = b*STEP+s (contiguous). f16 out.
// ---------------------------------------------------------------------------
__global__ __launch_bounds__(256)
void k_layernorm(const float* __restrict__ preds, const float* __restrict__ g,
                 const float* __restrict__ bta, _Float16* __restrict__ out) {
  int row  = blockIdx.x * 8 + (threadIdx.x >> 5);
  int lane = threadIdx.x & 31;
  const float* p = preds + (long)row * D_;
  float vals[16];
  float s = 0.f, s2 = 0.f;
  #pragma unroll
  for (int i = 0; i < 16; ++i) {
    float v = p[lane + i * 32];
    vals[i] = v; s += v; s2 += v * v;
  }
  #pragma unroll
  for (int off = 16; off > 0; off >>= 1) {
    s  += __shfl_xor(s,  off, 32);
    s2 += __shfl_xor(s2, off, 32);
  }
  float mu  = s * (1.f / D_);
  float var = s2 * (1.f / D_) - mu * mu;
  float inv = rsqrtf(var + EPS_);
  _Float16* op = out + (long)row * D_;
  #pragma unroll
  for (int i = 0; i < 16; ++i) {
    int c = lane + i * 32;
    op[c] = (_Float16)((vals[i] - mu) * inv * g[c] + bta[c]);
  }
}

// ---------------------------------------------------------------------------
// Scatter-accumulate decoder output into d_out.
// dec f32 [B*STEP, FIN] rows b*4+s, cols p*32+c -> out[b][(s*64+p)*32+c]
// mode 0: write; 1: add; 2: add then *std + mean (final block)
// ---------------------------------------------------------------------------
__global__ void k_accum_out(const float* __restrict__ dec, float* __restrict__ out,
                            const float* __restrict__ stdv, const float* __restrict__ meanv,
                            int mode) {
  long i = (long)blockIdx.x * 256 + threadIdx.x;    // over 1024*2048
  if (i >= (long)(B_ * STEP_) * FIN_) return;
  int col = (int)(i & (FIN_ - 1)), row = (int)(i >> 11);
  int b = row >> 2, s = row & 3;
  int p = col >> 5, c = col & 31;
  long oi = (long)b * (STEP_ * SEG_ * C_) + (long)(s * SEG_ + p) * C_ + c;
  float v = dec[i];
  if (mode == 0) {
    out[oi] = v;
  } else if (mode == 1) {
    out[oi] += v;
  } else {
    float acc = out[oi] + v;
    out[oi] = acc * stdv[b * C_ + c] + meanv[b * C_ + c];
  }
}

// ---------------------------------------------------------------------------
// Host orchestration
// ---------------------------------------------------------------------------
static inline char* ws_take(char*& p, size_t bytes) {
  char* r = p;
  p += (bytes + 255) & ~(size_t)255;
  return r;
}

extern "C" void kernel_launch(void* const* d_in, const int* in_sizes, int n_in,
                              void* d_out, int out_size, void* d_ws, size_t ws_size,
                              hipStream_t stream) {
  (void)in_sizes; (void)n_in; (void)out_size; (void)ws_size;

  const float* x_enc = (const float*)d_in[0];
  const float* maskw = (const float*)d_in[4];
  const float* ew1   = (const float*)d_in[5];
  const float* eb1   = (const float*)d_in[6];
  const float* ew2   = (const float*)d_in[7];
  const float* eb2   = (const float*)d_in[8];
  const float* wxh   = (const float*)d_in[9];
  const float* whh   = (const float*)d_in[10];
  const float* lng   = (const float*)d_in[11];
  const float* lnb   = (const float*)d_in[12];
  const float* dw1   = (const float*)d_in[13];
  const float* db1   = (const float*)d_in[14];
  const float* dw2   = (const float*)d_in[15];
  const float* db2   = (const float*)d_in[16];
  float* out = (float*)d_out;

  char* wp = (char*)d_ws;
  float*    meanv  = (float*)   ws_take(wp, BC_ * 4);
  float*    stdv   = (float*)   ws_take(wp, BC_ * 4);
  _Float16* Xn2    = (_Float16*)ws_take(wp, (size_t)BC_ * L_ * 2);          // [BC, L]
  float*    Kc     = (float*)   ws_take(wp, NB_ * L_ * 4);
  _Float16* Mc     = (_Float16*)ws_take(wp, (size_t)NB_ * L_ * L_ * 2);     // [NB,L,L]
  float*    Xi     = (float*)   ws_take(wp, (size_t)L_ * BC_ * 4);          // [L, BC]
  _Float16* res16  = (_Float16*)ws_take(wp, (size_t)B_ * FREQ_ * FIN_ * 2); // [2048,2048]
  _Float16* ew1h   = (_Float16*)ws_take(wp, (size_t)H_ * FIN_ * 2);
  _Float16* ew2h   = (_Float16*)ws_take(wp, (size_t)D_ * H_ * 2);
  _Float16* wxhh   = (_Float16*)ws_take(wp, (size_t)D_ * D_ * 2);
  _Float16* whhh   = (_Float16*)ws_take(wp, (size_t)D_ * D_ * 2);
  _Float16* dw1h   = (_Float16*)ws_take(wp, (size_t)H_ * D_ * 2);
  _Float16* dw2h   = (_Float16*)ws_take(wp, (size_t)FIN_ * H_ * 2);
  _Float16* e1h    = (_Float16*)ws_take(wp, (size_t)B_ * FREQ_ * H_ * 2);   // [2048,1024]
  _Float16* ench   = (_Float16*)ws_take(wp, (size_t)B_ * FREQ_ * D_ * 2);   // [2048,512]
  float*    xw     = (float*)   ws_take(wp, (size_t)B_ * FREQ_ * D_ * 4);   // [2048,512]
  _Float16* hA     = (_Float16*)ws_take(wp, (size_t)B_ * D_ * 2);
  _Float16* hB     = (_Float16*)ws_take(wp, (size_t)B_ * D_ * 2);
  float*    preds  = (float*)   ws_take(wp, (size_t)B_ * STEP_ * D_ * 4);   // [1024,512]
  _Float16* ln16   = (_Float16*)ws_take(wp, (size_t)B_ * STEP_ * D_ * 2);
  _Float16* d1h    = (_Float16*)ws_take(wp, (size_t)B_ * STEP_ * H_ * 2);   // [1024,1024]
  float*    d2f    = (float*)   ws_take(wp, (size_t)B_ * STEP_ * FIN_ * 4); // [1024,2048]

  auto gemm = [&](const _Float16* A, int lda, const _Float16* Bw, int ldb,
                  float* Cf, int ldc, _Float16* C16, int ldc16,
                  const float* bias, const float* aux, long auxld,
                  int M, int N, int K, int relu) {
    dim3 grid(N / 256, M / 128);
    k_gemm_wmma<<<grid, 256, 0, stream>>>(A, lda, Bw, ldb, Cf, ldc, C16, ldc16,
                                          bias, aux, auxld, K, relu);
  };
  auto cvt = [&](const float* s, _Float16* d, long n) {
    k_cvt16<<<(int)((n + 255) / 256), 256, 0, stream>>>(s, d, n);
  };

  // Stage 0: instance norm + filter construction
  k_instnorm<<<BC_ / 8, 256, 0, stream>>>(x_enc, meanv, stdv, Xn2);
  k_build_filter<<<(NB_ * L_ + 255) / 256, 256, 0, stream>>>(maskw, Kc);
  k_build_circ<<<(NB_ * L_ * L_) / 256, 256, 0, stream>>>(Kc, Mc);

  for (int nb = 0; nb < NB_; ++nb) {
    // Per-block weight conversion to f16
    cvt(ew1 + (size_t)nb * H_ * FIN_, ew1h, (long)H_ * FIN_);
    cvt(ew2 + (size_t)nb * D_ * H_,   ew2h, (long)D_ * H_);
    cvt(wxh + (size_t)nb * D_ * D_,   wxhh, (long)D_ * D_);
    cvt(whh + (size_t)nb * D_ * D_,   whhh, (long)D_ * D_);
    cvt(dw1 + (size_t)nb * H_ * D_,   dw1h, (long)H_ * D_);
    cvt(dw2 + (size_t)nb * FIN_ * H_, dw2h, (long)FIN_ * H_);

    // Fourier filter as circulant GEMM: Xi[L, BC] = Mc_nb[L,L] @ Xn2[BC,L]^T
    gemm(Mc + (size_t)nb * L_ * L_, L_, Xn2, L_, Xi, BC_, nullptr, 0,
         nullptr, nullptr, 0, L_, BC_, L_, 0);
    k_pack_res<<<(B_ * FREQ_ * FIN_) / 256, 256, 0, stream>>>(Xi, res16);

    // Encoder MLP
    gemm(res16, FIN_, ew1h, FIN_, nullptr, 0, e1h, H_,
         eb1 + nb * H_, nullptr, 0, B_ * FREQ_, H_, FIN_, 1);
    gemm(e1h, H_, ew2h, H_, nullptr, 0, ench, D_,
         eb2 + nb * D_, nullptr, 0, B_ * FREQ_, D_, H_, 0);

    // xw[b*8+f, :] = enc @ wxh^T  (f32, feeds recurrence epilogue adds)
    gemm(ench, D_, wxhh, D_, xw, D_, nullptr, 0,
         nullptr, nullptr, 0, B_ * FREQ_, D_, D_, 0);

    // h_1 = xw[:, f=0, :]
    k_cvt16_rows512<<<(B_ * D_) / 256, 256, 0, stream>>>(xw, hA, (long)B_ * D_,
                                                         (long)FREQ_ * D_);
    _Float16* cur = hA; _Float16* nxt = hB;
    // h_{f+1} = h_f @ whh^T + xw[:, f, :]   for f = 1..7
    for (int f = 1; f < FREQ_; ++f) {
      gemm(cur, D_, whhh, D_, nullptr, 0, nxt, D_,
           nullptr, xw + (size_t)f * D_, (long)FREQ_ * D_, B_, D_, D_, 0);
      _Float16* t = cur; cur = nxt; nxt = t;
    }
    // Free-run: preds[:, s, :] = h @ whh^T, h <- result, s = 0..3
    for (int s = 0; s < STEP_; ++s) {
      gemm(cur, D_, whhh, D_, preds + (size_t)s * D_, STEP_ * D_, nxt, D_,
           nullptr, nullptr, 0, B_, D_, D_, 0);
      _Float16* t = cur; cur = nxt; nxt = t;
    }

    // LayerNorm + decoder MLP
    k_layernorm<<<(B_ * STEP_) / 8, 256, 0, stream>>>(preds, lng + nb * D_,
                                                      lnb + nb * D_, ln16);
    gemm(ln16, D_, dw1h, D_, nullptr, 0, d1h, H_,
         db1 + nb * H_, nullptr, 0, B_ * STEP_, H_, D_, 1);
    gemm(d1h, H_, dw2h, H_, d2f, FIN_, nullptr, 0,
         db2 + nb * FIN_, nullptr, 0, B_ * STEP_, FIN_, H_, 0);

    // Accumulate into output; final block applies *std + mean
    int mode = (nb == 0) ? 0 : (nb == NB_ - 1 ? 2 : 1);
    k_accum_out<<<(B_ * STEP_ * FIN_) / 256, 256, 0, stream>>>(d2f, out, stdv, meanv, mode);
  }
}
